// WGCN_83803401880364
// MI455X (gfx1250) — matrix operand order repaired and checked by reading, not compiled, not run on hardware
//
#include <hip/hip_runtime.h>
#include <hip/hip_bf16.h>
#include <stdint.h>

#define N_NODES 50000
#define N_EDGES 800000
#define N_IN    96
#define N_HID   96
#define N_CLS   16

typedef __attribute__((ext_vector_type(16))) __bf16 v16bf;
typedef __attribute__((ext_vector_type(8)))  float  v8f;

// ---------------- helpers ----------------

__device__ __forceinline__ unsigned short f2bf_rne(float f) {
    unsigned int u = __float_as_uint(f);
    unsigned int r = (u + 0x7FFFu + ((u >> 16) & 1u)) >> 16;
    return (unsigned short)r;
}

__device__ __forceinline__ unsigned int mix_hash(unsigned int x) {
    x ^= x >> 16; x *= 0x7feb352du;
    x ^= x >> 15; x *= 0x846ca68bu;
    x ^= x >> 16;
    return x;
}

// ---------------- kernels ----------------

__global__ void k_zero_f32(float* __restrict__ p, int n) {
    int i = blockIdx.x * blockDim.x + threadIdx.x;
    if (i < n) p[i] = 0.0f;
}

__global__ void k_degree(const int* __restrict__ src, const int* __restrict__ dst,
                         float* __restrict__ deg_out, float* __restrict__ deg_in, int nE) {
    int e = blockIdx.x * blockDim.x + threadIdx.x;
    if (e >= nE) return;
    atomicAdd(deg_out + src[e], 1.0f);
    atomicAdd(deg_in  + dst[e], 1.0f);
}

__global__ void k_inv_sqrt_clamp(float* __restrict__ a, float* __restrict__ b, int n) {
    int i = blockIdx.x * blockDim.x + threadIdx.x;
    if (i >= n) return;
    a[i] = rsqrtf(fmaxf(a[i], 1.0f));
    b[i] = rsqrtf(fmaxf(b[i], 1.0f));
}

// dropout(p=0.5, scale 2x) fused with D_out^-1/2 row scaling
__global__ void k_drop_scale(const float* __restrict__ h, const float* __restrict__ inv_out,
                             float* __restrict__ xs, int n /* = N*96 */, unsigned int seed) {
    int i = blockIdx.x * blockDim.x + threadIdx.x;
    if (i >= n) return;
    int node = i / 96;
    unsigned int hsh = mix_hash((unsigned int)i * 0x9E3779B9u + seed);
    float v = (hsh >> 31) ? (h[i] * 2.0f) : 0.0f;
    xs[i] = v * inv_out[node];
}

// wave-per-edge gather + edge-weight scale + scatter-add (96 dims -> 3 per lane)
__global__ void k_edge_scatter(const float* __restrict__ xs, const float* __restrict__ ew,
                               const int* __restrict__ src, const int* __restrict__ dst,
                               float* __restrict__ agg, int nE) {
    int wave = (blockIdx.x * blockDim.x + threadIdx.x) >> 5;
    int lane = threadIdx.x & 31;
    if (wave >= nE) return;
    int s = src[wave];
    int d = dst[wave];
    float w = ew[wave];
    const float* xrow = xs + (size_t)s * 96;
    float*       arow = agg + (size_t)d * 96;
#pragma unroll
    for (int j = 0; j < 3; ++j) {
        int c = lane + 32 * j;
        atomicAdd(arow + c, xrow[c] * w);
    }
}

// D_in^-1/2 scaling fused with f32 -> bf16 (RNE) conversion for the WMMA A-matrix
__global__ void k_scale_tobf(const float* __restrict__ agg, const float* __restrict__ inv_in,
                             unsigned short* __restrict__ abf, int n /* = N*96 */) {
    int i = blockIdx.x * blockDim.x + threadIdx.x;
    if (i >= n) return;
    abf[i] = f2bf_rne(agg[i] * inv_in[i / 96]);
}

// Repack W [96 x NCOL] (row-major f32) into bf16 WMMA B-fragment order:
//   Wfrag[(kt*NT + nt)][lane][e]  contiguous, so each lane reads its whole
//   16-element fragment as two b128 loads in the GEMM inner loop.
// B layout per ISA: lane -> col (lane&15), elem e -> K = kt*32 + (lane>>4)*8 + (e&7) + (e>>3)*16.
__global__ void k_w_repack(const float* __restrict__ W, unsigned short* __restrict__ wf,
                           int NT, int NCOL, int total /* = 3*NT*512 */) {
    int i = blockIdx.x * blockDim.x + threadIdx.x;
    if (i >= total) return;
    int e    = i & 15;
    int lane = (i >> 4) & 31;
    int frag = i >> 9;            // kt*NT + nt
    int nt   = frag % NT;
    int kt   = frag / NT;
    int k = kt * 32 + (lane >> 4) * 8 + (e & 7) + (e >> 3) * 16;
    int n = nt * 16 + (lane & 15);
    wf[i] = f2bf_rne(W[(size_t)k * NCOL + n]);
}

// WMMA GEMM: out[M x (NT*16)] = A[M x 96](bf16) @ B[96 x (NT*16)](bf16, frag-packed) + bias.
// One wave computes a 16-row x (NT*16)-col tile via 3 k-steps of v_wmma_f32_16x16x32_bf16.
template <int NT, bool RELU>
__global__ void k_gemm_wmma(const unsigned short* __restrict__ Abf,
                            const unsigned short* __restrict__ Wfrag,
                            const float* __restrict__ bias,
                            float* __restrict__ out, int M) {
    const int wave = (blockIdx.x * blockDim.x + threadIdx.x) >> 5;
    const int lane = threadIdx.x & 31;
    const int tiles = M >> 4;
    if (wave >= tiles) return;

    const int half = lane >> 4;     // 0: lanes 0-15, 1: lanes 16-31
    const int lrow = lane & 15;
    const int arow = wave * 16 + lrow;
    const int NCOL = NT * 16;

    v8f acc[NT] = {};               // C/D layout: VGPR r -> row r + half*8, lane -> col

    union Frag { v16bf v; uint4 q[2]; };

#pragma unroll
    for (int kt = 0; kt < 3; ++kt) {
        // A 16x32 bf16 fragment: lane<16 holds K 0-7 & 16-23; lane>=16 holds K 8-15 & 24-31
        Frag a;
        const unsigned short* ap = Abf + (size_t)arow * 96 + kt * 32 + half * 8;
        a.q[0] = *(const uint4*)(ap);        // 8 contiguous bf16 (16B aligned)
        a.q[1] = *(const uint4*)(ap + 16);   // next 8 at K+16
#pragma unroll
        for (int nt = 0; nt < NT; ++nt) {
            // B fragment: pre-packed per (frag, lane); two contiguous b128 loads
            Frag b;
            const unsigned short* bp = Wfrag + ((size_t)(kt * NT + nt) * 32 + lane) * 16;
            b.q[0] = *(const uint4*)(bp);
            b.q[1] = *(const uint4*)(bp + 8);

            acc[nt] = __builtin_amdgcn_wmma_f32_16x16x32_bf16(
                /*neg_a=*/false, a.v, /*neg_b=*/false, b.v,
                /*c_mod=*/(short)0, acc[nt], /*reuse_a=*/false, /*reuse_b=*/false);
        }
    }

    // epilogue: bias (+ ReLU), scatter per documented C/D VGPR layout
#pragma unroll
    for (int nt = 0; nt < NT; ++nt) {
        const int col = nt * 16 + lrow;
        const float bb = bias[col];
#pragma unroll
        for (int r = 0; r < 8; ++r) {
            const int orow = wave * 16 + r + half * 8;
            float v = acc[nt][r] + bb;
            if (RELU) v = fmaxf(v, 0.0f);
            out[(size_t)orow * NCOL + col] = v;
        }
    }
}

__global__ void k_log_softmax16(const float* __restrict__ logits, float* __restrict__ out, int n) {
    int r = blockIdx.x * blockDim.x + threadIdx.x;
    if (r >= n) return;
    const float* x = logits + (size_t)r * 16;
    float m = x[0];
#pragma unroll
    for (int c = 1; c < 16; ++c) m = fmaxf(m, x[c]);
    float s = 0.0f;
#pragma unroll
    for (int c = 0; c < 16; ++c) s += expf(x[c] - m);
    float l = m + logf(s);
    float* o = out + (size_t)r * 16;
#pragma unroll
    for (int c = 0; c < 16; ++c) o[c] = x[c] - l;
}

// ---------------- launch ----------------

extern "C" void kernel_launch(void* const* d_in, const int* in_sizes, int n_in,
                              void* d_out, int out_size, void* d_ws, size_t ws_size,
                              hipStream_t stream) {
    (void)in_sizes; (void)n_in; (void)out_size; (void)ws_size;

    const float* feat = (const float*)d_in[0];
    const float* ew   = (const float*)d_in[1];
    const int*   src  = (const int*)  d_in[2];
    const int*   dst  = (const int*)  d_in[3];
    const float* W0   = (const float*)d_in[4];
    const float* b0   = (const float*)d_in[5];
    const float* W1   = (const float*)d_in[6];
    const float* b1   = (const float*)d_in[7];
    const float* W2   = (const float*)d_in[8];
    const float* b2   = (const float*)d_in[9];
    float* outp = (float*)d_out;

    const int N = N_NODES, E = N_EDGES;
    const int ND = N * 96;

    // scratch carving (256B aligned)
    char* ws = (char*)d_ws;
    size_t off = 0;
    auto carve = [&](size_t bytes) -> char* {
        char* p = ws + off;
        off = (off + bytes + 255) & ~(size_t)255;
        return p;
    };
    float*          inv_out = (float*)         carve((size_t)N * 4);
    float*          inv_in  = (float*)         carve((size_t)N * 4);
    float*          bufXS   = (float*)         carve((size_t)ND * 4);  // dropped+scaled h
    float*          bufAgg  = (float*)         carve((size_t)ND * 4);  // scatter target
    unsigned short* bufAbf  = (unsigned short*)carve((size_t)ND * 2);  // bf16 A-matrix
    float*          bufH    = (float*)         carve((size_t)ND * 4);  // layer output
    float*          bufC    = (float*)         carve((size_t)N * 16 * 4);
    unsigned short* bufWbf  = (unsigned short*)carve((size_t)96 * 96 * 2); // frag-packed W

    const int B = 256;
    auto g1 = [&](int n) { return (n + B - 1) / B; };

    // degrees -> inv-sqrt
    k_zero_f32<<<g1(N), B, 0, stream>>>(inv_out, N);
    k_zero_f32<<<g1(N), B, 0, stream>>>(inv_in, N);
    k_degree<<<g1(E), B, 0, stream>>>(src, dst, inv_out, inv_in, E);
    k_inv_sqrt_clamp<<<g1(N), B, 0, stream>>>(inv_out, inv_in, N);

    const int edge_blocks = (E * 32 + B - 1) / B;   // wave-per-edge
    const int tiles = N / 16;                       // 3125 exactly
    const int gemm_blocks = (tiles * 32 + B - 1) / B;

    const float* Ws[3] = {W0, W1, W2};
    const float* bs[3] = {b0, b1, b2};

    for (int layer = 0; layer < 3; ++layer) {
        const float* hin = (layer == 0) ? feat : bufH;
        const int NT = (layer < 2) ? 6 : 1;
        const int wtotal = 3 * NT * 512;            // = 96 * NT*16

        k_drop_scale<<<g1(ND), B, 0, stream>>>(hin, inv_out, bufXS, ND,
                                               0xC0FFEEu + (unsigned)layer * 0x85EBCA6Bu);
        k_zero_f32<<<g1(ND), B, 0, stream>>>(bufAgg, ND);
        k_edge_scatter<<<edge_blocks, B, 0, stream>>>(bufXS, ew, src, dst, bufAgg, E);
        k_scale_tobf<<<g1(ND), B, 0, stream>>>(bufAgg, inv_in, bufAbf, ND);
        k_w_repack<<<g1(wtotal), B, 0, stream>>>(Ws[layer], bufWbf, NT, NT * 16, wtotal);

        if (layer < 2) {
            k_gemm_wmma<6, true><<<gemm_blocks, B, 0, stream>>>(bufAbf, bufWbf, bs[layer], bufH, N);
        } else {
            k_gemm_wmma<1, false><<<gemm_blocks, B, 0, stream>>>(bufAbf, bufWbf, bs[layer], bufC, N);
        }
    }

    k_log_softmax16<<<g1(N), B, 0, stream>>>(bufC, outp, N);
}
